// LinformerSelfAttention_633
// MI455X (gfx1250) — matrix operand order repaired and checked by reading, not compiled, not run on hardware
//
#include <hip/hip_runtime.h>

// ---------------------------------------------------------------------------
// Linformer self-attention for MI455X (gfx1250, wave32, WMMA bf16 path).
// - All GEMMs: v_wmma_f32_16x16x32_bf16 (f32 acc), B operands transposed in
//   memory and swizzled [col][32k] in LDS -> all frag loads are ds_load_b128.
// - Global->LDS staging uses gfx1250 async-to-LDS ops (ASYNCcnt) with
//   double-buffered tiles in the two big GEMMs.
// ---------------------------------------------------------------------------

typedef __attribute__((ext_vector_type(16))) __bf16 v16bf;
typedef __attribute__((ext_vector_type(8)))  __bf16 v8bf;
typedef __attribute__((ext_vector_type(8)))  float  v8f;

constexpr int Bn = 8;
constexpr int Ln = 4096;
constexpr int Dn = 1024;
constexpr int Kn = 256;
constexpr int N3 = 3072;         // 3*Dn
constexpr int BL = Bn * Ln;      // 32768

// ---- async global->LDS (gfx1250) with safe fallback ------------------------
typedef __attribute__((__vector_size__(16))) int asy4i;
typedef __attribute__((__vector_size__(8)))  int asy2i;
#define AS1_ __attribute__((address_space(1)))
#define AS3_ __attribute__((address_space(3)))

#if __has_builtin(__builtin_amdgcn_global_load_async_to_lds_b128) && \
    __has_builtin(__builtin_amdgcn_global_load_async_to_lds_b64) &&  \
    __has_builtin(__builtin_amdgcn_s_wait_asynccnt)
#define ASYNC_CP16(dst, src)                                          \
  __builtin_amdgcn_global_load_async_to_lds_b128(                     \
      (AS1_ asy4i*)(void*)(src), (AS3_ asy4i*)(void*)(dst), 0, 0)
#define ASYNC_CP8(dst, src)                                           \
  __builtin_amdgcn_global_load_async_to_lds_b64(                      \
      (AS1_ asy2i*)(void*)(src), (AS3_ asy2i*)(void*)(dst), 0, 0)
#define ASYNC_WAIT() __builtin_amdgcn_s_wait_asynccnt(0)
#else
#define ASYNC_CP16(dst, src) (*(uint4*)(dst) = *(const uint4*)(src))
#define ASYNC_CP8(dst, src)  (*(uint2*)(dst) = *(const uint2*)(src))
#define ASYNC_WAIT() ((void)0)
#endif

// ------------------------- helpers -----------------------------------------

__device__ inline __bf16 f2bf(float f) {
  unsigned u = __builtin_bit_cast(unsigned, f);
  unsigned r = u + 0x7FFFu + ((u >> 16) & 1u);   // round-to-nearest-even
  unsigned short h = (unsigned short)(r >> 16);
  return __builtin_bit_cast(__bf16, h);
}

// A fragment (16x32 bf16, row-major source, leading dim ld).
// Lanes 0-15: M=0..15, K=0..15; lanes 16-31: K=16..31.
__device__ inline v16bf load_frag_a(const __bf16* p, int ld) {
  const int lane = threadIdx.x & 31;
  const int row  = lane & 15;
  const int kb   = (lane & 16) ? 16 : 0;
  const __bf16* rp = p + row * ld + kb;
  v8bf lo = *(const v8bf*)(rp);
  v8bf hi = *(const v8bf*)(rp + 8);
  return __builtin_shufflevector(lo, hi, 0, 1, 2, 3, 4, 5, 6, 7,
                                 8, 9, 10, 11, 12, 13, 14, 15);
}

// B fragment (32k x 16col) from swizzled LDS panel: element (k,c) at p[c*32+k].
__device__ inline v16bf load_frag_b_sw(const __bf16* p) {
  const int lane = threadIdx.x & 31;
  const int col  = lane & 15;
  const int kb   = (lane & 16) ? 16 : 0;
  return *(const v16bf*)(p + col * 32 + kb);
}

// C/D fragment (16x16 f32): lane<16 rows 0..7, lane>=16 rows 8..15, col=lane%16.
__device__ inline void store_frag_f32(float* p, int ld, v8f c) {
  const int lane = threadIdx.x & 31;
  const int col  = lane & 15;
  const int rb   = (lane & 16) ? 8 : 0;
#pragma unroll
  for (int v = 0; v < 8; ++v) p[(rb + v) * ld + col] = c[v];
}

__device__ inline void store_frag_bf16(__bf16* p, int ld, v8f c) {
  const int lane = threadIdx.x & 31;
  const int col  = lane & 15;
  const int rb   = (lane & 16) ? 8 : 0;
#pragma unroll
  for (int v = 0; v < 8; ++v) p[(size_t)(rb + v) * ld + col] = f2bf(c[v]);
}

// Transposed store: element (row rb+v, col) -> p[col*ld + rb+v]
__device__ inline void store_frag_bf16_T(__bf16* p, int ld, v8f c) {
  const int lane = threadIdx.x & 31;
  const int col  = lane & 15;
  const int rb   = (lane & 16) ? 8 : 0;
#pragma unroll
  for (int v = 0; v < 8; ++v) p[(size_t)col * ld + rb + v] = f2bf(c[v]);
}

#define WMMA_BF16(ACC, A, Bm) \
  (ACC) = __builtin_amdgcn_wmma_f32_16x16x32_bf16(false, (A), false, (Bm), (short)0, (ACC), false, false)

// ------------------------- conversion kernels ------------------------------

__global__ void kcvt(const float* __restrict__ s, __bf16* __restrict__ d, size_t n) {
  size_t i = ((size_t)blockIdx.x * blockDim.x + threadIdx.x) * 4;
  if (i + 3 < n) {
    float4 f = *(const float4*)(s + i);
    d[i + 0] = f2bf(f.x);
    d[i + 1] = f2bf(f.y);
    d[i + 2] = f2bf(f.z);
    d[i + 3] = f2bf(f.w);
  }
}

// s: [R][C] f32 -> d: [C][R] bf16, R = 1<<logR (shift/mask, no int division)
__global__ void kcvtT(const float* __restrict__ s, __bf16* __restrict__ d,
                      int logR, int C) {
  size_t o = (size_t)blockIdx.x * blockDim.x + threadIdx.x;   // over R*C
  int c = (int)(o >> logR);
  int r = (int)(o & (((size_t)1 << logR) - 1));
  d[o] = f2bf(s[(size_t)r * C + c]);
}

// ------------------------- QKV GEMM ----------------------------------------
// [B*L,3072] = X[B*L,1024] @ W[1024,3072] + bias.
// q cols (n<1024)      -> Qbuf [B*L][1024]  (row-major, A-operand later)
// k cols (1024..2047)  -> KT   [1024][B*L]  (transposed, B-operand later)
// v cols (2048..3071)  -> VT   [1024][B*L]  (transposed, B-operand later)
__global__ __launch_bounds__(256) void kqkv(const __bf16* __restrict__ X,
                                            const __bf16* __restrict__ WT,
                                            const float*  __restrict__ bias,
                                            __bf16* __restrict__ Qbuf,
                                            __bf16* __restrict__ KVT) {
  __shared__ __align__(32) __bf16 As[2][128 * 32];
  __shared__ __align__(32) __bf16 Bs[2][128 * 32];   // swizzled: [col][32k]
  const int t   = threadIdx.x;
  const int wid = t >> 5;
  const int m0  = blockIdx.y * 128;
  const int n0  = blockIdx.x * 128;
  const int wm  = wid & 3;        // 4 row-groups of 32
  const int wn  = wid >> 2;       // 2 col-groups of 64
  v8f acc[2][4] = {};

  const int rT = t >> 1, hT = (t & 1) * 16;   // staging coords (shared A/B)
  auto stage = [&](int buf, int kk) {
    const __bf16* gA = X + (size_t)(m0 + rT) * Dn + kk + hT;
    __bf16* sA = As[buf] + rT * 32 + hT;
    ASYNC_CP16(sA, gA);
    ASYNC_CP16(sA + 8, gA + 8);
    const __bf16* gB = WT + (size_t)(n0 + rT) * Dn + kk + hT;
    __bf16* sB = Bs[buf] + rT * 32 + hT;
    ASYNC_CP16(sB, gB);
    ASYNC_CP16(sB + 8, gB + 8);
  };

  stage(0, 0);
  ASYNC_WAIT();
  __syncthreads();
  for (int kk = 0; kk < Dn; kk += 32) {
    const int cur = (kk >> 5) & 1;
    if (kk + 32 < Dn) stage(cur ^ 1, kk + 32);
    if (kk + 64 < Dn) {   // keep a speculative prefetch ahead of the pipeline
      __builtin_prefetch(X + (size_t)(m0 + rT) * Dn + kk + 64, 0, 1);
      __builtin_prefetch(WT + (size_t)(n0 + rT) * Dn + kk + 64, 0, 1);
    }
    v16bf a[2], b[4];
#pragma unroll
    for (int i = 0; i < 2; ++i) a[i] = load_frag_a(As[cur] + (wm * 32 + i * 16) * 32, 32);
#pragma unroll
    for (int j = 0; j < 4; ++j) b[j] = load_frag_b_sw(Bs[cur] + (wn * 64 + j * 16) * 32);
#pragma unroll
    for (int i = 0; i < 2; ++i)
#pragma unroll
      for (int j = 0; j < 4; ++j) WMMA_BF16(acc[i][j], a[i], b[j]);
    ASYNC_WAIT();
    __syncthreads();
  }

  const int lane = t & 31;
  const int col  = lane & 15;
#pragma unroll
  for (int i = 0; i < 2; ++i)
#pragma unroll
    for (int j = 0; j < 4; ++j) {
      const int r0 = m0 + wm * 32 + i * 16;
      const int c0 = n0 + wn * 64 + j * 16;
      const float bv = bias[c0 + col];
      v8f c = acc[i][j];
#pragma unroll
      for (int v = 0; v < 8; ++v) c[v] += bv;
      if (c0 < Dn) {
        store_frag_bf16(Qbuf + (size_t)r0 * Dn + c0, Dn, c);
      } else {
        __bf16* T = KVT + (size_t)((c0 >> 10) - 1) * Dn * BL;   // 0=K, 1=V
        store_frag_bf16_T(T + (size_t)(c0 & 1023) * BL + r0, BL, c);
      }
    }
}

// ------------------------- projection GEMMs --------------------------------
// z even: K_proj[b][kidx][d]  (row-major out)
// z odd : V_projT[b][d][kidx] (transposed out)
__global__ __launch_bounds__(256) void kproj(const __bf16* __restrict__ HkT,
                                             const __bf16* __restrict__ HvT,
                                             const __bf16* __restrict__ KVT,
                                             __bf16* __restrict__ Kp,
                                             __bf16* __restrict__ VpT) {
  __shared__ __align__(32) __bf16 As[2][128 * 32];
  __shared__ __align__(32) __bf16 Bs[2][128 * 32];   // swizzled: [d-col][32 l]
  const int z  = blockIdx.z;
  const int bb = z >> 1;
  const bool isV = z & 1;
  const __bf16* A   = isV ? HvT : HkT;                       // [256, 4096]
  const __bf16* Src = KVT + (size_t)(isV ? 1 : 0) * Dn * BL; // [Dn][B*L]
  const int t   = threadIdx.x;
  const int wid = t >> 5;
  const int m0  = blockIdx.y * 128;   // kidx
  const int n0  = blockIdx.x * 128;   // d
  const int wm  = wid & 3, wn = wid >> 2;
  v8f acc[2][4] = {};

  const int rT = t >> 1, hT = (t & 1) * 16;
  auto stage = [&](int buf, int ll) {
    const __bf16* gA = A + (size_t)(m0 + rT) * Ln + ll + hT;
    __bf16* sA = As[buf] + rT * 32 + hT;
    ASYNC_CP16(sA, gA);
    ASYNC_CP16(sA + 8, gA + 8);
    const __bf16* gB = Src + (size_t)(n0 + rT) * BL + (size_t)bb * Ln + ll + hT;
    __bf16* sB = Bs[buf] + rT * 32 + hT;
    ASYNC_CP16(sB, gB);
    ASYNC_CP16(sB + 8, gB + 8);
  };

  stage(0, 0);
  ASYNC_WAIT();
  __syncthreads();
  for (int ll = 0; ll < Ln; ll += 32) {
    const int cur = (ll >> 5) & 1;
    if (ll + 32 < Ln) stage(cur ^ 1, ll + 32);
    v16bf a[2], b[4];
#pragma unroll
    for (int i = 0; i < 2; ++i) a[i] = load_frag_a(As[cur] + (wm * 32 + i * 16) * 32, 32);
#pragma unroll
    for (int j = 0; j < 4; ++j) b[j] = load_frag_b_sw(Bs[cur] + (wn * 64 + j * 16) * 32);
#pragma unroll
    for (int i = 0; i < 2; ++i)
#pragma unroll
      for (int j = 0; j < 4; ++j) WMMA_BF16(acc[i][j], a[i], b[j]);
    ASYNC_WAIT();
    __syncthreads();
  }

#pragma unroll
  for (int i = 0; i < 2; ++i)
#pragma unroll
    for (int j = 0; j < 4; ++j) {
      const int r0 = m0 + wm * 32 + i * 16;   // kidx
      const int c0 = n0 + wn * 64 + j * 16;   // d
      if (isV)
        store_frag_bf16_T(VpT + (size_t)bb * Dn * Kn + (size_t)c0 * Kn + r0, Kn, acc[i][j]);
      else
        store_frag_bf16(Kp + (size_t)bb * Kn * Dn + (size_t)r0 * Dn + c0, Dn, acc[i][j]);
    }
}

// ------------------------- fused attention ---------------------------------
// Per block: 32 L-rows. scores[32,256] -> softmax -> out[32,1024] (f32).
__global__ __launch_bounds__(256) void kattn(const __bf16* __restrict__ Qbuf,
                                             const __bf16* __restrict__ Kp,
                                             const __bf16* __restrict__ VpT,
                                             float* __restrict__ Out) {
  __shared__ __align__(32) char smem[18432 + 32768];   // 50 KB, aliased phases
  __shared__ float rowm[32], rinv[32];
  __bf16* qs  = (__bf16*)smem;              // phase1: 32x32 row-major (2 KB)
  __bf16* kp  = (__bf16*)(smem + 2048);     // phase1: swizzled [kidx][32d] (16 KB)
  float*  sc  = (float*)(smem + 18432);     // scores: 32x256 (32 KB)
  __bf16* wts = (__bf16*)smem;              // phase2: 32x256 row-major (16 KB)
  __bf16* vs  = (__bf16*)(smem + 18432);    // phase2: swizzled [d][32kidx] (32 KB)

  const int t   = threadIdx.x;
  const int wid = t >> 5;
  const int bb  = blockIdx.x >> 7;          // Ln/32 = 128 tiles per batch
  const int l0  = (blockIdx.x & 127) * 32;
  const __bf16* Qb = Qbuf + (size_t)((size_t)bb * Ln + l0) * Dn;
  const __bf16* Kb = Kp  + (size_t)bb * Kn * Dn;   // [Kn][Dn]
  const __bf16* Vb = VpT + (size_t)bb * Dn * Kn;   // [Dn][Kn]

  // ---- phase 1: scores = q @ K_proj^T / sqrt(D) ----
  v8f accs[2][2] = {};
  for (int dd = 0; dd < Dn; dd += 32) {
    {  // q tile 32x32 row-major: 4 bf16 (8B) per thread
      ASYNC_CP8(qs + (t >> 3) * 32 + (t & 7) * 4,
                Qb + (size_t)(t >> 3) * Dn + dd + (t & 7) * 4);
    }
    {  // K_proj tile: 256 kidx cols x 32 d, swizzled; thread t owns col t
      const __bf16* g = Kb + (size_t)t * Dn + dd;
      __bf16* s = kp + t * 32;
      ASYNC_CP16(s, g);       ASYNC_CP16(s + 8, g + 8);
      ASYNC_CP16(s + 16, g + 16); ASYNC_CP16(s + 24, g + 24);
    }
    ASYNC_WAIT();
    __syncthreads();
    v16bf a[2], b[2];
#pragma unroll
    for (int i = 0; i < 2; ++i) a[i] = load_frag_a(qs + i * 16 * 32, 32);
#pragma unroll
    for (int j = 0; j < 2; ++j) b[j] = load_frag_b_sw(kp + (wid * 32 + j * 16) * 32);
#pragma unroll
    for (int i = 0; i < 2; ++i)
#pragma unroll
      for (int j = 0; j < 2; ++j) WMMA_BF16(accs[i][j], a[i], b[j]);
    __syncthreads();
  }
#pragma unroll
  for (int i = 0; i < 2; ++i)
#pragma unroll
    for (int j = 0; j < 2; ++j) {
      v8f c = accs[i][j];
#pragma unroll
      for (int v = 0; v < 8; ++v) c[v] *= 0.03125f;     // 1/sqrt(1024)
      store_frag_f32(sc + (i * 16) * 256 + wid * 32 + j * 16, 256, c);
    }
  __syncthreads();

  // ---- softmax over 256 columns ----
  if (t < 32) {
    float m = -3.4e38f;
    for (int j = 0; j < 256; ++j) m = fmaxf(m, sc[t * 256 + j]);
    float s = 0.f;
    for (int j = 0; j < 256; ++j) s += __expf(sc[t * 256 + j] - m);
    rowm[t] = m;
    rinv[t] = 1.f / s;
  }
  __syncthreads();
  {
    const int r = t >> 3, cb = (t & 7) * 32;
    const float m = rowm[r], s = rinv[r];
#pragma unroll
    for (int j = 0; j < 32; ++j)
      wts[r * 256 + cb + j] = f2bf(__expf(sc[r * 256 + cb + j] - m) * s);
  }
  __syncthreads();

  // ---- phase 2: out = weights @ V_proj ----
  for (int nh = 0; nh < 2; ++nh) {
    v8f acco[2][4] = {};
    for (int kc = 0; kc < Kn; kc += 32) {
      // V tile: 512 d cols x 32 kidx, swizzled; thread t owns cols 2t, 2t+1
#pragma unroll
      for (int u = 0; u < 2; ++u) {
        const int c = 2 * t + u;
        const __bf16* g = Vb + (size_t)(nh * 512 + c) * Kn + kc;
        __bf16* s = vs + c * 32;
        ASYNC_CP16(s, g);       ASYNC_CP16(s + 8, g + 8);
        ASYNC_CP16(s + 16, g + 16); ASYNC_CP16(s + 24, g + 24);
      }
      ASYNC_WAIT();
      __syncthreads();
      v16bf a[2], b[4];
#pragma unroll
      for (int i = 0; i < 2; ++i) a[i] = load_frag_a(wts + i * 16 * 256 + kc, 256);
#pragma unroll
      for (int j = 0; j < 4; ++j) b[j] = load_frag_b_sw(vs + (wid * 64 + j * 16) * 32);
#pragma unroll
      for (int i = 0; i < 2; ++i)
#pragma unroll
        for (int j = 0; j < 4; ++j) WMMA_BF16(acco[i][j], a[i], b[j]);
      __syncthreads();
    }
#pragma unroll
    for (int i = 0; i < 2; ++i)
#pragma unroll
      for (int j = 0; j < 4; ++j)
        store_frag_f32(Out + (size_t)((size_t)bb * Ln + l0 + i * 16) * Dn + nh * 512 + wid * 64 + j * 16,
                       Dn, acco[i][j]);
  }
}

// ------------------------- launcher ----------------------------------------

extern "C" void kernel_launch(void* const* d_in, const int* in_sizes, int n_in,
                              void* d_out, int out_size, void* d_ws, size_t ws_size,
                              hipStream_t stream) {
  (void)in_sizes; (void)n_in; (void)out_size; (void)ws_size;
  const float* x    = (const float*)d_in[0];
  const float* W    = (const float*)d_in[1];
  const float* bias = (const float*)d_in[2];
  const float* Hk   = (const float*)d_in[3];
  const float* Hv   = (const float*)d_in[4];

  char* ws = (char*)d_ws;
  size_t off = 0;
  __bf16* XB   = (__bf16*)(ws + off); off += (size_t)BL * Dn * 2;        //  64 MB
  __bf16* WT   = (__bf16*)(ws + off); off += (size_t)N3 * Dn * 2;        //   6 MB  [3072][1024]
  __bf16* HKT  = (__bf16*)(ws + off); off += (size_t)Kn * Ln * 2;        //   2 MB  [256][4096]
  __bf16* HVT  = (__bf16*)(ws + off); off += (size_t)Kn * Ln * 2;        //   2 MB
  __bf16* QBUF = (__bf16*)(ws + off); off += (size_t)BL * Dn * 2;        //  64 MB  [B*L][1024]
  __bf16* KVT  = (__bf16*)(ws + off); off += (size_t)2 * Dn * BL * 2;    // 128 MB  [2][1024][B*L]
  __bf16* KP   = (__bf16*)(ws + off); off += (size_t)Bn * Kn * Dn * 2;   //   4 MB  [B][256][1024]
  __bf16* VPT  = (__bf16*)(ws + off); off += (size_t)Bn * Dn * Kn * 2;   //   4 MB  [B][1024][256]

  kcvt<<<(int)(((size_t)BL * Dn) / 1024), 256, 0, stream>>>(x, XB, (size_t)BL * Dn);
  kcvtT<<<(N3 * Dn) / 256, 256, 0, stream>>>(W, WT, 10, N3);    // W[1024][3072] -> WT[3072][1024]
  kcvtT<<<(Kn * Ln) / 256, 256, 0, stream>>>(Hk, HKT, 12, Kn);  // Hk[4096][256] -> HKT[256][4096]
  kcvtT<<<(Kn * Ln) / 256, 256, 0, stream>>>(Hv, HVT, 12, Kn);

  kqkv<<<dim3(N3 / 128, BL / 128), 256, 0, stream>>>(XB, WT, bias, QBUF, KVT);
  kproj<<<dim3(Dn / 128, Kn / 128, Bn * 2), 256, 0, stream>>>(HKT, HVT, KVT, KP, VPT);
  kattn<<<Bn * (Ln / 32), 256, 0, stream>>>(QBUF, KP, VPT, (float*)d_out);
}
